// GNN_no_edge_attr_27187142983847
// MI455X (gfx1250) — matrix operand order repaired and checked by reading, not compiled, not run on hardware
//
#include <hip/hip_runtime.h>

#define NN      100000   // nodes
#define NE      1600000  // edges
#define NG      4096     // graphs
#define FDIM    128      // in/hid dim
#define ODIM    64       // out dim
#define BN_EPS  1e-5f

typedef __bf16 bf16_t;
typedef bf16_t v16bf __attribute__((ext_vector_type(16)));
typedef float  v8f   __attribute__((ext_vector_type(8)));

struct Frag32B { uint4 lo, hi; };   // 32 bytes == v16bf

static __device__ __forceinline__ unsigned short f2bf(float f) {
  unsigned int u = __float_as_uint(f);
  u += 0x7FFFu + ((u >> 16) & 1u);          // round-to-nearest-even
  return (unsigned short)(u >> 16);
}

static __device__ __forceinline__ void atomAddF(float* p, float v) {
  unsafeAtomicAdd(p, v);                    // hardware global_atomic_add_f32
}

// ---------------- small utility kernels ----------------
__global__ void k_fill(float* __restrict__ p, float v, int n) {
  int i = blockIdx.x * blockDim.x + threadIdx.x;
  if (i < n) p[i] = v;
}

__global__ void k_deg_accum(const int* __restrict__ dst, float* __restrict__ deg) {
  int e = blockIdx.x * blockDim.x + threadIdx.x;
  if (e < NE) atomAddF(&deg[dst[e]], 1.0f);
}

__global__ void k_rsqrt_inplace(float* __restrict__ p, int n) {
  int i = blockIdx.x * blockDim.x + threadIdx.x;
  if (i < n) p[i] = rsqrtf(p[i]);
}

__global__ void k_f32_to_bf16(const float* __restrict__ x, unsigned short* __restrict__ xb, int n) {
  int i = blockIdx.x * blockDim.x + threadIdx.x;
  if (i < n) xb[i] = f2bf(x[i]);
}

// ---------------- W packing into WMMA B-fragment order ----------------
// Wpack dword index: ((ct*4 + kt)*32 + lane)*8 + v
// lane: n = ct*16 + (lane&15), half h = lane>>4
// dword v holds bf16 pair (k, k+1) with k = kt*32 + h*8 + v*2 + (v>>2)*8
__global__ void k_packW(const float* __restrict__ W, unsigned int* __restrict__ Wp) {
  int idx = blockIdx.x * blockDim.x + threadIdx.x;   // 8*4*32*8 = 8192
  if (idx >= 8192) return;
  int v    = idx & 7;
  int lane = (idx >> 3) & 31;
  int kt   = (idx >> 8) & 3;
  int ct   = idx >> 10;
  int n = ct * 16 + (lane & 15);
  int k = kt * 32 + (lane >> 4) * 8 + v * 2 + (v >> 2) * 8;
  unsigned int lo = f2bf(W[(size_t)k * FDIM + n]);
  unsigned int hi = f2bf(W[(size_t)(k + 1) * FDIM + n]);
  Wp[idx] = lo | (hi << 16);
}

// ---------------- WMMA GEMM: C[NN,128] = Abf16[NN,128] @ W ----------------
// block = 256 threads = 8 waves; wave `ct` computes the 16x16 tile at
// rows [16*blockIdx.x, +16), cols [16*ct, +16). K=128 in 4 chunks of 32.
__global__ void k_gemm_bf16(const unsigned short* __restrict__ A,
                            const unsigned int*  __restrict__ Wp,
                            float* __restrict__ C) {
  int row0 = blockIdx.x * 16;
  int lane = threadIdx.x & 31;
  int ct   = threadIdx.x >> 5;          // 0..7 column tile
  int m    = lane & 15;
  int h    = lane >> 4;

  const unsigned short* arow = A + (size_t)(row0 + m) * FDIM + h * 8;
  const uint4* bbase = ((const uint4*)Wp) + ((size_t)ct * 4) * 64;  // 32 lanes * 2 uint4

  v8f acc = {};
#pragma unroll
  for (int kt = 0; kt < 4; ++kt) {
    Frag32B fa;
    fa.lo = *(const uint4*)(arow + kt * 32);        // K = kt*32 + h*8 + 0..7
    fa.hi = *(const uint4*)(arow + kt * 32 + 16);   // K = kt*32 + h*8 + 16..23
    const uint4* bp = bbase + ((size_t)kt * 32 + lane) * 2;
    Frag32B fb;
    fb.lo = bp[0];
    fb.hi = bp[1];
    v16bf a = __builtin_bit_cast(v16bf, fa);
    v16bf b = __builtin_bit_cast(v16bf, fb);
    acc = __builtin_amdgcn_wmma_f32_16x16x32_bf16(false, a, false, b,
                                                  (short)0, acc, false, false);
  }
  // D layout: VGPR r -> row (r + 8*h), col = lane&15
#pragma unroll
  for (int r = 0; r < 8; ++r) {
    C[(size_t)(row0 + r + 8 * h) * FDIM + ct * 16 + m] = acc[r];
  }
}

// ---------------- GCN aggregation ----------------
// agg = h * dinv^2 + bias   (self-loop term + bias)
__global__ void k_agg_init(const float* __restrict__ H, const float* __restrict__ dinv,
                           const float* __restrict__ bias, float* __restrict__ A) {
  int idx = blockIdx.x * blockDim.x + threadIdx.x;   // NN*128 exact
  int c = idx & 127;
  int i = idx >> 7;
  float di = dinv[i];
  A[idx] = H[idx] * di * di + bias[c];
}

// wave per edge: agg[dst] += h[src] * dinv[src]*dinv[dst]
__global__ void k_agg_edges(const int* __restrict__ src, const int* __restrict__ dst,
                            const float* __restrict__ dinv,
                            const float* __restrict__ H, float* __restrict__ A) {
  int w = (blockIdx.x * blockDim.x + threadIdx.x) >> 5;  // global wave = edge
  if (w >= NE) return;
  int lane = threadIdx.x & 31;
  int s = src[w];
  int d = dst[w];
  float norm = dinv[s] * dinv[d];
  float4 hv = *(const float4*)(H + (size_t)s * FDIM + lane * 4);
  float* ap = A + (size_t)d * FDIM + lane * 4;
  atomAddF(ap + 0, hv.x * norm);
  atomAddF(ap + 1, hv.y * norm);
  atomAddF(ap + 2, hv.z * norm);
  atomAddF(ap + 3, hv.w * norm);
}

// ---------------- BatchNorm ----------------
__global__ void k_bn_reduce(const float* __restrict__ A, float* __restrict__ sums) {
  int c = threadIdx.x;                 // 128 threads
  int r0 = blockIdx.x * 512;
  int r1 = r0 + 512; if (r1 > NN) r1 = NN;
  float s = 0.f, q = 0.f;
  for (int r = r0; r < r1; ++r) {
    float v = A[(size_t)r * FDIM + c];
    s += v; q += v * v;
  }
  atomAddF(&sums[c], s);
  atomAddF(&sums[FDIM + c], q);
}

__global__ void k_bn_final(const float* __restrict__ sums, const float* __restrict__ g,
                           const float* __restrict__ be, float* __restrict__ par) {
  int c = threadIdx.x;
  if (c < FDIM) {
    const float invN = 1.0f / (float)NN;
    float mu  = sums[c] * invN;
    float var = sums[FDIM + c] * invN - mu * mu;
    float sc  = g[c] * rsqrtf(var + BN_EPS);
    par[c]        = sc;
    par[FDIM + c] = be[c] - mu * sc;
  }
}

// fused BN + ReLU; emit bf16 (next GEMM input) and f32 (pooling input)
__global__ void k_bn_apply(const float* __restrict__ A, const float* __restrict__ par,
                           unsigned short* __restrict__ xb, float* __restrict__ Hout) {
  int idx = blockIdx.x * blockDim.x + threadIdx.x;   // NN*128 exact
  int c = idx & 127;
  float v = fmaf(A[idx], par[c], par[FDIM + c]);
  v = fmaxf(v, 0.0f);
  xb[idx]   = f2bf(v);
  Hout[idx] = v;
}

// ---------------- pooling + FC ----------------
__global__ void k_pool(const float* __restrict__ H, const int* __restrict__ batch,
                       float* __restrict__ pooled, float* __restrict__ cnt) {
  int w = (blockIdx.x * blockDim.x + threadIdx.x) >> 5;  // wave = node
  if (w >= NN) return;
  int lane = threadIdx.x & 31;
  int g = batch[w];
  float4 hv = *(const float4*)(H + (size_t)w * FDIM + lane * 4);
  float* pp = pooled + (size_t)g * FDIM + lane * 4;
  atomAddF(pp + 0, hv.x);
  atomAddF(pp + 1, hv.y);
  atomAddF(pp + 2, hv.z);
  atomAddF(pp + 3, hv.w);
  if (lane == 0) atomAddF(&cnt[g], 1.0f);
}

__global__ void k_fc(const float* __restrict__ pooled, const float* __restrict__ cnt,
                     const float* __restrict__ W, const float* __restrict__ b,
                     float* __restrict__ out) {
  int idx = blockIdx.x * blockDim.x + threadIdx.x;   // NG*64
  if (idx >= NG * ODIM) return;
  int g = idx >> 6;
  int o = idx & 63;
  float inv = 1.0f / fmaxf(cnt[g], 1.0f);
  const float* pr = pooled + (size_t)g * FDIM;
  float acc = 0.f;
#pragma unroll 8
  for (int k = 0; k < FDIM; ++k) acc = fmaf(pr[k], W[(size_t)k * ODIM + o], acc);
  out[idx] = fmaf(acc, inv, b[o]);
}

// ---------------- host launch ----------------
extern "C" void kernel_launch(void* const* d_in, const int* in_sizes, int n_in,
                              void* d_out, int out_size, void* d_ws, size_t ws_size,
                              hipStream_t stream) {
  (void)in_sizes; (void)n_in; (void)out_size; (void)ws_size;

  const float* x     = (const float*)d_in[0];
  const int*   eidx  = (const int*)d_in[1];
  const int*   batch = (const int*)d_in[2];
  const float* Ws[3]   = {(const float*)d_in[3],  (const float*)d_in[7],  (const float*)d_in[11]};
  const float* bs[3]   = {(const float*)d_in[4],  (const float*)d_in[8],  (const float*)d_in[12]};
  const float* gs[3]   = {(const float*)d_in[5],  (const float*)d_in[9],  (const float*)d_in[13]};
  const float* bes[3]  = {(const float*)d_in[6],  (const float*)d_in[10], (const float*)d_in[14]};
  const float* fcW = (const float*)d_in[15];
  const float* fcb = (const float*)d_in[16];
  float* out = (float*)d_out;

  const int* src = eidx;
  const int* dst = eidx + NE;

  // workspace carve-up (256B aligned)
  char* base = (char*)d_ws;
  size_t off = 0;
  auto carve = [&](size_t bytes) -> char* {
    char* p = base + off;
    off = (off + bytes + 255) & ~(size_t)255;
    return p;
  };
  float*          bufH   = (float*)         carve((size_t)NN * FDIM * 4);  // GEMM out / post-BN f32
  float*          bufA   = (float*)         carve((size_t)NN * FDIM * 4);  // aggregation
  unsigned short* xbf    = (unsigned short*)carve((size_t)NN * FDIM * 2);  // bf16 GEMM input
  float*          dinv   = (float*)         carve((size_t)NN * 4);
  unsigned int*   Wpack  = (unsigned int*)  carve(8192 * 4);
  float*          bnsums = (float*)         carve(256 * 4);
  float*          bnpar  = (float*)         carve(256 * 4);
  float*          pooled = (float*)         carve((size_t)NG * FDIM * 4);
  float*          cnt    = (float*)         carve((size_t)NG * 4);

  const int T = 256;
  const int nodeElems = NN * FDIM;          // 12,800,000 = 50000 * 256

  // degree -> dinv (shared by all layers)
  k_fill<<<(NN + T - 1) / T, T, 0, stream>>>(dinv, 1.0f, NN);          // +1 self loop
  k_deg_accum<<<(NE + T - 1) / T, T, 0, stream>>>(dst, dinv);
  k_rsqrt_inplace<<<(NN + T - 1) / T, T, 0, stream>>>(dinv, NN);

  // layer-0 input -> bf16
  k_f32_to_bf16<<<nodeElems / T, T, 0, stream>>>(x, xbf, nodeElems);

  for (int l = 0; l < 3; ++l) {
    k_packW<<<8192 / T, T, 0, stream>>>(Ws[l], Wpack);
    k_gemm_bf16<<<NN / 16, T, 0, stream>>>(xbf, Wpack, bufH);
    k_agg_init<<<nodeElems / T, T, 0, stream>>>(bufH, dinv, bs[l], bufA);
    k_agg_edges<<<(NE * 32 + T - 1) / T, T, 0, stream>>>(src, dst, dinv, bufH, bufA);
    k_fill<<<1, T, 0, stream>>>(bnsums, 0.0f, 256);
    k_bn_reduce<<<(NN + 511) / 512, FDIM, 0, stream>>>(bufA, bnsums);
    k_bn_final<<<1, FDIM, 0, stream>>>(bnsums, gs[l], bes[l], bnpar);
    k_bn_apply<<<nodeElems / T, T, 0, stream>>>(bufA, bnpar, xbf, bufH);
  }

  // global mean pool + FC
  k_fill<<<(NG * FDIM + T - 1) / T, T, 0, stream>>>(pooled, 0.0f, NG * FDIM);
  k_fill<<<(NG + T - 1) / T, T, 0, stream>>>(cnt, 0.0f, NG);
  k_pool<<<(NN * 32 + T - 1) / T, T, 0, stream>>>(bufH, batch, pooled, cnt);
  k_fc<<<(NG * ODIM) / T, T, 0, stream>>>(pooled, cnt, fcW, fcb, out);
}
